// AnomalyDAE_4544075399675
// MI455X (gfx1250) — matrix operand order repaired and checked by reading, not compiled, not run on hardware
//
#include <hip/hip_runtime.h>

typedef __attribute__((ext_vector_type(16))) __bf16 v16bf;
typedef __attribute__((ext_vector_type(8)))  float  v8f;

#define IN_DIM 512
#define EMBED  128
#define OUT_F  64
#define LDSPITCH 520   // 512 + 8 bf16 pad: lane banks = m*4 mod 64 -> conflict-free

// Monotone order-preserving float<->uint key for atomicMax on floats.
static __device__ __forceinline__ unsigned f2key(float f) {
    unsigned u = __float_as_uint(f);
    return (u & 0x80000000u) ? ~u : (u | 0x80000000u);
}
static __device__ __forceinline__ float key2f(unsigned k) {
    return __uint_as_float((k & 0x80000000u) ? (k ^ 0x80000000u) : ~k);
}

// ---- 1) convert+transpose weights to bf16 -----------------------------------
__global__ void prep_weights(const float* __restrict__ Wd, const float* __restrict__ Wg,
                             __bf16* __restrict__ Wt, __bf16* __restrict__ Wgt) {
    int i = blockIdx.x * 256 + threadIdx.x;
    if (i < EMBED * IN_DIM) {               // Wt[k][o] = W_dense[o][k]
        int o = i / IN_DIM, k = i % IN_DIM;
        Wt[k * EMBED + o] = (__bf16)Wd[i];
    }
    int j = i - EMBED * IN_DIM;
    if (j >= 0 && j < OUT_F * EMBED) {      // Wgt[k][o] = W_gat[o][k]
        int o = j / EMBED, k = j % EMBED;
        Wgt[k * OUT_F + o] = (__bf16)Wg[j];
    }
}

// ---- 2) init out=b_gat, segmax=key(-inf)=0, segsum=0 ------------------------
__global__ void init_out(const float* __restrict__ bg, float* __restrict__ out,
                         unsigned* __restrict__ segmax, float* __restrict__ segsum, int N) {
    int i = blockIdx.x * 256 + threadIdx.x;
    if (i < N * OUT_F) out[i] = bg[i & (OUT_F - 1)];
    if (i < N) { segmax[i] = 0u; segsum[i] = 0.0f; }
}

// ---- 3) h = leakyReLU(x @ Wdense^T + b), bf16 WMMA, f32 accum ---------------
// Block = 256 threads (8 waves), 32 rows x 128 cols. Whole 32x512 A panel is
// staged to LDS once (one barrier); each wave owns one 16-col tile and carries
// two accumulators (rows 0-15 / 16-31) so every B fragment feeds two WMMAs.
__global__ __launch_bounds__(256) void gemm1(const float* __restrict__ x,
                                             const __bf16* __restrict__ Wt,
                                             const float* __restrict__ bias,
                                             __bf16* __restrict__ h, int N) {
    __shared__ __bf16 lds_a[32 * LDSPITCH];
    const int t    = threadIdx.x;
    const int wave = t >> 5;           // 8 waves -> 8 column tiles of 16 (EMBED=128)
    const int lane = t & 31;
    const int m    = lane & 15;
    const int kh   = (lane >> 4) << 3; // K sub-group per A-layout (0 or 8)
    const int rowBase = blockIdx.x * 32;

    // stage 32x512 f32 -> bf16 (float4 reads, packed bf16 LDS writes)
    for (int i = t * 4; i < 32 * IN_DIM; i += 256 * 4) {
        int r = i >> 9, k = i & (IN_DIM - 1);
        int rr = rowBase + r; if (rr >= N) rr = N - 1;
        const float4 v = *(const float4*)(x + (size_t)rr * IN_DIM + k);
        __bf16* lp = &lds_a[r * LDSPITCH + k];
        lp[0] = (__bf16)v.x; lp[1] = (__bf16)v.y;
        lp[2] = (__bf16)v.z; lp[3] = (__bf16)v.w;
    }
    __syncthreads();

    v8f c0 = {}, c1 = {};
    for (int kk = 0; kk < IN_DIM; kk += 32) {
        // A fragments: lane row m (and m+16), K = kh..kh+7 and 16+kh..+7
        v16bf a0, a1, b;
        const __bf16* ap0 = &lds_a[m * LDSPITCH + kk + kh];
        const __bf16* ap1 = &lds_a[(m + 16) * LDSPITCH + kk + kh];
        #pragma unroll
        for (int i = 0; i < 8; ++i) {
            a0[i] = ap0[i]; a0[8 + i] = ap0[16 + i];
            a1[i] = ap1[i]; a1[8 + i] = ap1[16 + i];
        }
        // B fragment: lane col n = m, K = kh..kh+7 and 16+kh..+7 from transposed W
        const __bf16* bp = Wt + (size_t)(kk + kh) * EMBED + wave * 16 + m;
        #pragma unroll
        for (int jj = 0; jj < 8; ++jj) { b[jj] = bp[jj * EMBED]; b[8 + jj] = bp[(16 + jj) * EMBED]; }
        c0 = __builtin_amdgcn_wmma_f32_16x16x32_bf16(false, a0, false, b, (short)0, c0, false, false);
        c1 = __builtin_amdgcn_wmma_f32_16x16x32_bf16(false, a1, false, b, (short)0, c1, false, false);
    }

    // C/D layout: lane -> col n = lane&15, rows M = (lane>>4)*8 + r
    const int col  = wave * 16 + m;
    const float bc = bias[col];
    #pragma unroll
    for (int r2 = 0; r2 < 8; ++r2) {
        int row0 = rowBase + kh + r2;
        int row1 = row0 + 16;
        float v0 = c0[r2] + bc; v0 = v0 > 0.0f ? v0 : 0.01f * v0;
        float v1 = c1[r2] + bc; v1 = v1 > 0.0f ? v1 : 0.01f * v1;
        if (row0 < N) h[(size_t)row0 * EMBED + col] = (__bf16)v0;
        if (row1 < N) h[(size_t)row1 * EMBED + col] = (__bf16)v1;
    }
}

// ---- 4) g = h @ Wgat^T (bf16 WMMA -> f32) -----------------------------------
__global__ __launch_bounds__(256) void gemm2(const __bf16* __restrict__ h,
                                             const __bf16* __restrict__ Wgt,
                                             float* __restrict__ g, int N) {
    const int t    = threadIdx.x;
    const int wave = t >> 5;
    const int lane = t & 31;
    const int rw   = wave >> 2;        // 2 row tiles per block
    const int cw   = wave & 3;         // 4 col tiles (OUT_F=64)
    const int rowTile = blockIdx.x * 2 + rw;
    const int numRowTiles = (N + 15) / 16;
    if (rowTile >= numRowTiles) return;   // wave-uniform: EXEC stays all-1s
    const int m  = lane & 15;
    const int kh = (lane >> 4) << 3;
    const int rowBase = rowTile * 16;
    v8f c = {};
    for (int kk = 0; kk < EMBED; kk += 32) {
        v16bf a, b;
        int row = rowBase + m; if (row >= N) row = N - 1;
        const __bf16* ap = h + (size_t)row * EMBED + kk + kh;
        #pragma unroll
        for (int i = 0; i < 8; ++i) { a[i] = ap[i]; a[8 + i] = ap[16 + i]; }
        const __bf16* bp = Wgt + (size_t)(kk + kh) * OUT_F + cw * 16 + m;
        #pragma unroll
        for (int jj = 0; jj < 8; ++jj) { b[jj] = bp[jj * OUT_F]; b[8 + jj] = bp[(16 + jj) * OUT_F]; }
        c = __builtin_amdgcn_wmma_f32_16x16x32_bf16(false, a, false, b, (short)0, c, false, false);
    }
    const int col = cw * 16 + m;
    #pragma unroll
    for (int r2 = 0; r2 < 8; ++r2) {
        int row = rowBase + kh + r2;
        if (row < N) g[(size_t)row * OUT_F + col] = c[r2];
    }
}

// ---- 5) per-node attention scalars ------------------------------------------
__global__ void att_kernel(const float* __restrict__ g, const float* __restrict__ as_,
                           const float* __restrict__ ad_, float* __restrict__ a_src,
                           float* __restrict__ a_dst, int N) {
    int i = blockIdx.x * 256 + threadIdx.x;
    if (i >= N) return;
    const float* gp = g + (size_t)i * OUT_F;
    float s = 0.0f, d = 0.0f;
    #pragma unroll
    for (int f = 0; f < OUT_F; ++f) { float gv = gp[f]; s += gv * as_[f]; d += gv * ad_[f]; }
    a_src[i] = s; a_dst[i] = d;
}

// ---- 6) segment max over destinations ---------------------------------------
__global__ void edge_max(const int* __restrict__ ei, const float* __restrict__ a_src,
                         const float* __restrict__ a_dst, unsigned* __restrict__ segmax,
                         int E, int N) {
    int t = blockIdx.x * 256 + threadIdx.x;
    if (t >= E + N) return;
    int s, d;
    if (t < E) { s = ei[t]; d = ei[E + t]; } else { s = d = t - E; }
    float e = a_src[s] + a_dst[d];
    e = e > 0.0f ? e : 0.2f * e;
    atomicMax(&segmax[d], f2key(e));
}

// ---- 7) exp(e - max) + segment sum ------------------------------------------
__global__ void edge_exp(const int* __restrict__ ei, const float* __restrict__ a_src,
                         const float* __restrict__ a_dst, const unsigned* __restrict__ segmax,
                         float* __restrict__ segsum, float* __restrict__ wedge, int E, int N) {
    int t = blockIdx.x * 256 + threadIdx.x;
    if (t >= E + N) return;
    int s, d;
    if (t < E) { s = ei[t]; d = ei[E + t]; } else { s = d = t - E; }
    float e = a_src[s] + a_dst[d];
    e = e > 0.0f ? e : 0.2f * e;
    float w = expf(e - key2f(segmax[d]));
    wedge[t] = w;
    atomicAdd(&segsum[d], w);
}

// ---- 8) weighted scatter-add: one wave per edge -----------------------------
__global__ __launch_bounds__(256) void aggregate(const int* __restrict__ ei,
                                                 const float* __restrict__ g,
                                                 const float* __restrict__ wedge,
                                                 const float* __restrict__ segsum,
                                                 float* __restrict__ out, int E, int N) {
    int eid = blockIdx.x * 8 + (threadIdx.x >> 5);
    if (eid >= E + N) return;
    int lane = threadIdx.x & 31;
    int s, d;
    if (eid < E) { s = ei[eid]; d = ei[E + eid]; } else { s = d = eid - E; }
    float alpha = wedge[eid] / segsum[d];
    const float* gp = g + (size_t)s * OUT_F;
    float* op = out + (size_t)d * OUT_F;
    atomicAdd(&op[lane],      alpha * gp[lane]);
    atomicAdd(&op[lane + 32], alpha * gp[lane + 32]);
}

extern "C" void kernel_launch(void* const* d_in, const int* in_sizes, int n_in,
                              void* d_out, int out_size, void* d_ws, size_t ws_size,
                              hipStream_t stream) {
    const float* x     = (const float*)d_in[0];
    const int*   ei    = (const int*)  d_in[1];
    const float* Wd    = (const float*)d_in[2];
    const float* bd    = (const float*)d_in[3];
    const float* Wg    = (const float*)d_in[4];
    const float* att_s = (const float*)d_in[5];
    const float* att_d = (const float*)d_in[6];
    const float* bg    = (const float*)d_in[7];
    float* out = (float*)d_out;
    const int N = in_sizes[0] / IN_DIM;
    const int E = in_sizes[1] / 2;

    char* p = (char*)d_ws;
    auto alloc = [&](size_t bytes) -> char* {
        char* q = p; p += (bytes + 255) & ~(size_t)255; return q;
    };
    __bf16*   Wt    = (__bf16*)  alloc((size_t)IN_DIM * EMBED * 2);
    __bf16*   Wgt   = (__bf16*)  alloc((size_t)EMBED * OUT_F * 2);
    __bf16*   hbuf  = (__bf16*)  alloc((size_t)N * EMBED * 2);
    float*    gbuf  = (float*)   alloc((size_t)N * OUT_F * 4);
    float*    asrc  = (float*)   alloc((size_t)N * 4);
    float*    adst  = (float*)   alloc((size_t)N * 4);
    unsigned* smax  = (unsigned*)alloc((size_t)N * 4);
    float*    ssum  = (float*)   alloc((size_t)N * 4);
    float*    wedge = (float*)   alloc((size_t)(E + N) * 4);

    prep_weights<<<(EMBED * IN_DIM + OUT_F * EMBED + 255) / 256, 256, 0, stream>>>(Wd, Wg, Wt, Wgt);
    init_out<<<((size_t)N * OUT_F + 255) / 256, 256, 0, stream>>>(bg, out, smax, ssum, N);
    gemm1<<<(N + 31) / 32, 256, 0, stream>>>(x, Wt, bd, hbuf, N);
    int rt = (N + 15) / 16;
    gemm2<<<(rt + 1) / 2, 256, 0, stream>>>(hbuf, Wgt, gbuf, N);
    att_kernel<<<(N + 255) / 256, 256, 0, stream>>>(gbuf, att_s, att_d, asrc, adst, N);
    int tot = E + N;
    edge_max<<<(tot + 255) / 256, 256, 0, stream>>>(ei, asrc, adst, smax, E, N);
    edge_exp<<<(tot + 255) / 256, 256, 0, stream>>>(ei, asrc, adst, smax, ssum, wedge, E, N);
    aggregate<<<(tot + 7) / 8, 256, 0, stream>>>(ei, gbuf, wedge, ssum, out, E, N);
}
